// SpatiotemporalAttentionSingleDir_4020089389354
// MI455X (gfx1250) — matrix operand order repaired and checked by reading, not compiled, not run on hardware
//
#include <hip/hip_runtime.h>

typedef __bf16 bf16_t;
typedef __attribute__((ext_vector_type(16))) __bf16 v16bf;
typedef __attribute__((ext_vector_type(8)))  float  v8f;

#define kB  64
#define kF  2048
#define kV  90
#define kVp 96
#define kI  1024
#define kEps 1e-5f

// ---------------- WMMA fragment helpers (wave32, 16x16x32 bf16) ----------------

__device__ __forceinline__ v8f wmma_bf16(v16bf a, v16bf b, v8f c) {
  return __builtin_amdgcn_wmma_f32_16x16x32_bf16(false, a, false, b, (short)0, c,
                                                 false, false);
}

// A fragment 16x32 (MxK). Lanes 0-15: M=lane, K 0..15; lanes 16-31: M=lane-16,
// K 16..31 (matches B). trans=false: A[m,k] = base[m*ld + k]; trans=true: base[k*ld + m].
__device__ __forceinline__ v16bf load_frag_a(const bf16_t* base, int ld,
                                             int m0, int k0, bool trans) {
  int lane = threadIdx.x & 31;
  int m = m0 + (lane & 15);
  int kb = k0 + ((lane >> 4) << 4);  // +16 for upper half
  v16bf f;
#pragma unroll
  for (int t = 0; t < 8; ++t) {
    int k = kb + 2 * t;
    if (!trans) {
      f[2 * t]     = base[(size_t)m * ld + k];
      f[2 * t + 1] = base[(size_t)m * ld + k + 1];
    } else {
      f[2 * t]     = base[(size_t)k * ld + m];
      f[2 * t + 1] = base[(size_t)(k + 1) * ld + m];
    }
  }
  return f;
}

// B fragment 32x16 (KxN), direct from global.
// Lanes 0-15: N=lane, K 0..15; lanes 16-31: N=lane-16, K 16..31.
// trans=false: B[k,n] = base[k*ld + n]; trans=true: B[k,n] = base[n*ld + k].
__device__ __forceinline__ v16bf load_frag_b(const bf16_t* base, int ld,
                                             int k0, int n0, bool trans) {
  int lane = threadIdx.x & 31;
  int n = n0 + (lane & 15);
  int kb = k0 + ((lane >> 4) << 4);
  v16bf f;
#pragma unroll
  for (int t = 0; t < 8; ++t) {
    int k = kb + 2 * t;
    if (!trans) {
      f[2 * t]     = base[(size_t)k * ld + n];
      f[2 * t + 1] = base[(size_t)(k + 1) * ld + n];
    } else {
      f[2 * t]     = base[(size_t)n * ld + k];
      f[2 * t + 1] = base[(size_t)n * ld + k + 1];
    }
  }
  return f;
}

// ---- LDS-staged B panel: fragment-ready layout, one panel per block ----
// Element B[k, n0+n] lives at sB[(ktile*32 + lane)*16 + pos] with
// lane = ((k&31)>=16 ? 16:0) | n, pos = k & 15. Each wave lane then reads its
// whole fragment as 32 contiguous bytes (ds_load_b128 x2), shared by all waves.
template <int K>
__device__ __forceinline__ void stage_b_panel(const bf16_t* __restrict__ Bsrc,
                                              int n0,
                                              unsigned short* __restrict__ sB) {
  const unsigned short* src = (const unsigned short*)Bsrc;
  int n2 = (threadIdx.x & 7) << 1;
#pragma unroll 4
  for (int k = (int)(threadIdx.x >> 3); k < K; k += 16) {
    unsigned v = *(const unsigned*)(src + (size_t)k * kVp + n0 + n2);
    int tile = k >> 5, kt = k & 31, pos = kt & 15, lh = (kt >> 4) << 4;
    sB[((tile * 32 + (lh | n2)) << 4) | pos]       = (unsigned short)(v & 0xffffu);
    sB[((tile * 32 + (lh | (n2 + 1))) << 4) | pos] = (unsigned short)(v >> 16);
  }
}

__device__ __forceinline__ v16bf load_frag_b_lds(const unsigned short* sB,
                                                 int tile) {
  int lane = threadIdx.x & 31;
  return *(const v16bf*)(sB + (((size_t)tile * 32 + lane) << 4));
}

// C/D layout: lanes 0-15: N=lane, VGPR r holds M=m0+r; lanes 16-31: M=m0+8+r.
__device__ __forceinline__ void store_tile_bf16(bf16_t* base, int ld,
                                                int m0, int n0, v8f c) {
  int lane = threadIdx.x & 31;
  int n = n0 + (lane & 15);
  int mr = m0 + ((lane >> 4) << 3);
#pragma unroll
  for (int r = 0; r < 8; ++r) base[(size_t)(mr + r) * ld + n] = (bf16_t)c[r];
}

// ---------------- small prep kernels ----------------

__global__ void k_fold_w(const float* __restrict__ w, const float* __restrict__ bn,
                         bf16_t* __restrict__ wout, int O, int C) {
  int idx = blockIdx.x * blockDim.x + threadIdx.x;
  if (idx >= O * C) return;
  int f = idx % C;
  float scale = bn[f] * rsqrtf(bn[3 * C + f] + kEps);
  wout[idx] = (bf16_t)(w[idx] * scale);
}

__global__ void k_eff_bias(const float* __restrict__ w, const float* __restrict__ bn,
                           const float* __restrict__ b, float* __restrict__ bout,
                           int C) {
  __shared__ float red[256];
  int o = blockIdx.x;
  float s = 0.f;
  for (int f = threadIdx.x; f < C; f += 256) {
    float scale = bn[f] * rsqrtf(bn[3 * C + f] + kEps);
    s += w[(size_t)o * C + f] * (bn[C + f] - bn[2 * C + f] * scale);
  }
  red[threadIdx.x] = s;
  __syncthreads();
  for (int st = 128; st > 0; st >>= 1) {
    if (threadIdx.x < st) red[threadIdx.x] += red[threadIdx.x + st];
    __syncthreads();
  }
  if (threadIdx.x == 0) bout[o] = b[o] + red[0];
}

__global__ void k_cast_bf16(const float* __restrict__ in, bf16_t* __restrict__ out,
                            int n) {
  int idx = blockIdx.x * blockDim.x + threadIdx.x;
  if (idx < n) out[idx] = (bf16_t)in[idx];
}

__global__ void k_cast_pad(const float* __restrict__ x, bf16_t* __restrict__ out) {
  int idx = blockIdx.x * blockDim.x + threadIdx.x;
  if (idx >= kB * kF * kVp) return;
  int v = idx % kVp;
  int bf = idx / kVp;
  out[idx] = (v < kV) ? (bf16_t)x[(size_t)bf * kV + v] : (bf16_t)0.f;
}

__global__ void k_mean_v(const float* __restrict__ x, float* __restrict__ xm) {
  int idx = blockIdx.x * blockDim.x + threadIdx.x;
  if (idx >= kB * kF) return;
  const float* p = x + (size_t)idx * kV;
  float s = 0.f;
  for (int v = 0; v < kV; ++v) s += p[v];
  xm[idx] = s * (1.0f / kV);
}

__global__ void k_copy(const float* __restrict__ in, float* __restrict__ out,
                       size_t n) {
  size_t idx = (size_t)blockIdx.x * blockDim.x + threadIdx.x;
  if (idx < n) out[idx] = in[idx];
}

// ---------------- WMMA GEMM kernels ----------------

// Projection: out[b] = Wbf(I,F) @ Xbf[b](F,Vp) + bias -> bf16 [B,I,Vp], pad zeroed.
// grid (I/64, Vp/16, B), block 128 (4 waves). 64KB LDS B panel.
__global__ void __launch_bounds__(128) k_proj_gemm(const bf16_t* __restrict__ W,
                                                   const float* __restrict__ bias,
                                                   const bf16_t* __restrict__ X,
                                                   bf16_t* __restrict__ out) {
  __shared__ __align__(32) unsigned short sB[kF * 16];
  int b = blockIdx.z;
  int wv = threadIdx.x >> 5;
  int m0 = blockIdx.x * 64 + wv * 16;
  int n0 = blockIdx.y * 16;
  stage_b_panel<kF>(X + (size_t)b * kF * kVp, n0, sB);
  __syncthreads();
  v8f acc = {};
  for (int t = 0; t < kF / 32; ++t) {
    if (t + 2 < kF / 32)
      __builtin_prefetch(W + (size_t)m0 * kF + (t + 2) * 32, 0, 1);
    acc = wmma_bf16(load_frag_a(W, kF, m0, t * 32, false),
                    load_frag_b_lds(sB, t), acc);
  }
  int lane = threadIdx.x & 31;
  int n = n0 + (lane & 15);
  int mr = m0 + ((lane >> 4) << 3);
  bf16_t* ob = out + (size_t)b * kI * kVp;
#pragma unroll
  for (int r = 0; r < 8; ++r) {
    float val = (n < kV) ? (acc[r] + bias[mr + r]) : 0.f;
    ob[(size_t)(mr + r) * kVp + n] = (bf16_t)val;
  }
}

// energy_time rows + fused softmax: E[i,j] = sum_v theta[i,v]*phi[j,v]; softmax j.
// grid (I/16, B), block 256 (8 waves), 64KB LDS.
__global__ void __launch_bounds__(256) k_energy_time(const bf16_t* __restrict__ theta,
                                                     const bf16_t* __restrict__ phi,
                                                     bf16_t* __restrict__ attn) {
  __shared__ float E[16 * kI];
  int b = blockIdx.y;
  int m0 = blockIdx.x * 16;
  const bf16_t* Tb = theta + (size_t)b * kI * kVp;
  const bf16_t* Pb = phi + (size_t)b * kI * kVp;
  int wv = threadIdx.x >> 5, lane = threadIdx.x & 31;
  for (int jt = wv; jt < kI / 16; jt += 8) {
    int n0 = jt * 16;
    v8f acc = {};
#pragma unroll
    for (int k0 = 0; k0 < kVp; k0 += 32) {
      acc = wmma_bf16(load_frag_a(Tb, kVp, m0, k0, false),
                      load_frag_b(Pb, kVp, k0, n0, true), acc);  // B = phi^T
    }
    int n = n0 + (lane & 15);
    int mr = (lane >> 4) << 3;
#pragma unroll
    for (int r = 0; r < 8; ++r) E[(mr + r) * kI + n] = acc[r];
  }
  __syncthreads();
  bf16_t* Ab = attn + (size_t)b * kI * kI + (size_t)m0 * kI;
  for (int r = wv; r < 16; r += 8) {
    float mx = -3.4e38f;
    for (int c = lane; c < kI; c += 32) mx = fmaxf(mx, E[r * kI + c]);
    for (int off = 16; off; off >>= 1) mx = fmaxf(mx, __shfl_xor(mx, off, 32));
    float sum = 0.f;
    for (int c = lane; c < kI; c += 32) {
      float e = __expf(E[r * kI + c] - mx);
      E[r * kI + c] = e;
      sum += e;
    }
    for (int off = 16; off; off >>= 1) sum += __shfl_xor(sum, off, 32);
    float inv = 1.f / sum;
    for (int c = lane; c < kI; c += 32)
      Ab[(size_t)r * kI + c] = (bf16_t)(E[r * kI + c] * inv);
  }
}

// energy_space + fused softmax: E2[u,v] = sum_i phi[i,u]*theta[i,v]; softmax v<90.
// grid (B), block 256 (8 waves), 36KB LDS.
__global__ void __launch_bounds__(256) k_energy_space(const bf16_t* __restrict__ theta,
                                                      const bf16_t* __restrict__ phi,
                                                      bf16_t* __restrict__ attn) {
  __shared__ float E[kVp * kVp];
  int b = blockIdx.x;
  const bf16_t* Tb = theta + (size_t)b * kI * kVp;
  const bf16_t* Pb = phi + (size_t)b * kI * kVp;
  int wv = threadIdx.x >> 5, lane = threadIdx.x & 31;
  for (int t = wv; t < 36; t += 8) {
    int m0 = (t / 6) * 16, n0 = (t % 6) * 16;
    v8f acc = {};
    for (int k0 = 0; k0 < kI; k0 += 32) {
      acc = wmma_bf16(load_frag_a(Pb, kVp, m0, k0, true),   // A = phi^T (u x i)
                      load_frag_b(Tb, kVp, k0, n0, false), acc);
    }
    int n = n0 + (lane & 15);
    int mr = m0 + ((lane >> 4) << 3);
#pragma unroll
    for (int r = 0; r < 8; ++r) E[(mr + r) * kVp + n] = acc[r];
  }
  __syncthreads();
  bf16_t* Ab = attn + (size_t)b * kVp * kVp;
  for (int u = wv; u < kVp; u += 8) {
    if (u < kV) {
      float mx = -3.4e38f;
      for (int c = lane; c < kV; c += 32) mx = fmaxf(mx, E[u * kVp + c]);
      for (int off = 16; off; off >>= 1) mx = fmaxf(mx, __shfl_xor(mx, off, 32));
      float sum = 0.f;
      for (int c = lane; c < kV; c += 32) {
        float e = __expf(E[u * kVp + c] - mx);
        E[u * kVp + c] = e;
        sum += e;
      }
      for (int off = 16; off; off >>= 1) sum += __shfl_xor(sum, off, 32);
      float inv = 1.f / sum;
      for (int c = lane; c < kVp; c += 32)
        Ab[u * kVp + c] = (bf16_t)((c < kV) ? E[u * kVp + c] * inv : 0.f);
    } else {
      for (int c = lane; c < kVp; c += 32) Ab[u * kVp + c] = (bf16_t)0.f;
    }
  }
}

// y[b] = attn_time[b](I,I) @ g[b](I,Vp). grid (I/64, Vp/16, B), block 128. 32KB LDS.
__global__ void __launch_bounds__(128) k_gemm_at_g(const bf16_t* __restrict__ attn,
                                                   const bf16_t* __restrict__ g,
                                                   bf16_t* __restrict__ y) {
  __shared__ __align__(32) unsigned short sB[kI * 16];
  int b = blockIdx.z;
  int wv = threadIdx.x >> 5;
  int m0 = blockIdx.x * 64 + wv * 16;
  int n0 = blockIdx.y * 16;
  const bf16_t* A = attn + (size_t)b * kI * kI;
  stage_b_panel<kI>(g + (size_t)b * kI * kVp, n0, sB);
  __syncthreads();
  v8f acc = {};
  for (int t = 0; t < kI / 32; ++t) {
    if (t + 2 < kI / 32)
      __builtin_prefetch(A + (size_t)m0 * kI + (t + 2) * 32, 0, 1);
    acc = wmma_bf16(load_frag_a(A, kI, m0, t * 32, false),
                    load_frag_b_lds(sB, t), acc);
  }
  store_tile_bf16(y + (size_t)b * kI * kVp, kVp, m0, n0, acc);
}

// z2[b] = y[b](I,Vp) @ attn_space[b](Vp,Vp). grid (I/64, Vp/16, B), block 128.
__global__ void __launch_bounds__(128) k_gemm_z2(const bf16_t* __restrict__ y,
                                                 const bf16_t* __restrict__ attn,
                                                 bf16_t* __restrict__ z2) {
  __shared__ __align__(32) unsigned short sB[kVp * 16];
  int b = blockIdx.z;
  int wv = threadIdx.x >> 5;
  int m0 = blockIdx.x * 64 + wv * 16;
  int n0 = blockIdx.y * 16;
  const bf16_t* Yb = y + (size_t)b * kI * kVp;
  stage_b_panel<kVp>(attn + (size_t)b * kVp * kVp, n0, sB);
  __syncthreads();
  v8f acc = {};
#pragma unroll
  for (int t = 0; t < kVp / 32; ++t)
    acc = wmma_bf16(load_frag_a(Yb, kVp, m0, t * 32, false),
                    load_frag_b_lds(sB, t), acc);
  store_tile_bf16(z2 + (size_t)b * kI * kVp, kVp, m0, n0, acc);
}

// out2 = x2 + x1_mean + BN_W(W_w @ z2 + W_b). grid (F/64, Vp/16, B), block 128.
__global__ void __launch_bounds__(128) k_final(const bf16_t* __restrict__ Wbf,
                                               const bf16_t* __restrict__ z2,
                                               const float* __restrict__ Wb,
                                               const float* __restrict__ Wbn,
                                               const float* __restrict__ x2,
                                               const float* __restrict__ x1m,
                                               float* __restrict__ out2) {
  __shared__ __align__(32) unsigned short sB[kI * 16];
  int b = blockIdx.z;
  int wv = threadIdx.x >> 5;
  int m0 = blockIdx.x * 64 + wv * 16;
  int n0 = blockIdx.y * 16;
  stage_b_panel<kI>(z2 + (size_t)b * kI * kVp, n0, sB);
  __syncthreads();
  v8f acc = {};
  for (int t = 0; t < kI / 32; ++t) {
    if (t + 2 < kI / 32)
      __builtin_prefetch(Wbf + (size_t)m0 * kI + (t + 2) * 32, 0, 1);
    acc = wmma_bf16(load_frag_a(Wbf, kI, m0, t * 32, false),
                    load_frag_b_lds(sB, t), acc);
  }
  int lane = threadIdx.x & 31;
  int n = n0 + (lane & 15);
  if (n >= kV) return;
  int mr = m0 + ((lane >> 4) << 3);
#pragma unroll
  for (int r = 0; r < 8; ++r) {
    int f = mr + r;
    float scale = Wbn[f] * rsqrtf(Wbn[3 * kF + f] + kEps);
    float val = acc[r] + Wb[f];
    val = (val - Wbn[2 * kF + f]) * scale + Wbn[kF + f];
    size_t o = ((size_t)b * kF + f) * kV + n;
    out2[o] = x2[o] + x1m[(size_t)b * kF + f] + val;
  }
}

// ---------------- host launch ----------------

extern "C" void kernel_launch(void* const* d_in, const int* in_sizes, int n_in,
                              void* d_out, int out_size, void* d_ws, size_t ws_size,
                              hipStream_t stream) {
  (void)in_sizes; (void)n_in; (void)out_size; (void)ws_size;
  const float* x1    = (const float*)d_in[0];
  const float* x2    = (const float*)d_in[1];
  const float* g_bn  = (const float*)d_in[2];
  const float* g_w   = (const float*)d_in[3];
  const float* g_b   = (const float*)d_in[4];
  const float* th_bn = (const float*)d_in[5];
  const float* th_w  = (const float*)d_in[6];
  const float* th_b  = (const float*)d_in[7];
  const float* ph_bn = (const float*)d_in[8];
  const float* ph_w  = (const float*)d_in[9];
  const float* ph_b  = (const float*)d_in[10];
  const float* W_w   = (const float*)d_in[11];
  const float* W_b   = (const float*)d_in[12];
  const float* W_bn  = (const float*)d_in[13];
  // qt_w/qt_b/ks_w/ks_b (14..17) are dead: softmax over a singleton axis == 1.

  float* out1 = (float*)d_out;
  float* out2 = out1 + (size_t)kB * kF * kV;

  char* ws = (char*)d_ws;
  size_t off = 0;
  auto take = [&](size_t bytes) -> char* {
    char* p = ws + off;
    off += (bytes + 255) & ~(size_t)255;
    return p;
  };

  bf16_t* x1bf = (bf16_t*)take((size_t)kB * kF * kVp * 2);
  bf16_t* x2bf = (bf16_t*)take((size_t)kB * kF * kVp * 2);
  bf16_t* wg   = (bf16_t*)take((size_t)kI * kF * 2);
  bf16_t* wth  = (bf16_t*)take((size_t)kI * kF * 2);
  bf16_t* wph  = (bf16_t*)take((size_t)kI * kF * 2);
  bf16_t* wW   = (bf16_t*)take((size_t)kF * kI * 2);
  float*  bg   = (float*)take((size_t)kI * 4);
  float*  bth  = (float*)take((size_t)kI * 4);
  float*  bph  = (float*)take((size_t)kI * 4);
  bf16_t* gb   = (bf16_t*)take((size_t)kB * kI * kVp * 2);
  bf16_t* thb  = (bf16_t*)take((size_t)kB * kI * kVp * 2);
  bf16_t* phb  = (bf16_t*)take((size_t)kB * kI * kVp * 2);
  bf16_t* at   = (bf16_t*)take((size_t)kB * kI * kI * 2);
  bf16_t* as   = (bf16_t*)take((size_t)kB * kVp * kVp * 2);
  bf16_t* ybf  = (bf16_t*)take((size_t)kB * kI * kVp * 2);
  bf16_t* z2bf = (bf16_t*)take((size_t)kB * kI * kVp * 2);
  float*  x1m  = (float*)take((size_t)kB * kF * 4);

  const int nWF = kI * kF;
  // weight prep
  k_fold_w<<<(nWF + 255) / 256, 256, 0, stream>>>(g_w, g_bn, wg, kI, kF);
  k_fold_w<<<(nWF + 255) / 256, 256, 0, stream>>>(th_w, th_bn, wth, kI, kF);
  k_fold_w<<<(nWF + 255) / 256, 256, 0, stream>>>(ph_w, ph_bn, wph, kI, kF);
  k_eff_bias<<<kI, 256, 0, stream>>>(g_w, g_bn, g_b, bg, kF);
  k_eff_bias<<<kI, 256, 0, stream>>>(th_w, th_bn, th_b, bth, kF);
  k_eff_bias<<<kI, 256, 0, stream>>>(ph_w, ph_bn, ph_b, bph, kF);
  k_cast_bf16<<<(kF * kI + 255) / 256, 256, 0, stream>>>(W_w, wW, kF * kI);
  // input prep
  const int nXP = kB * kF * kVp;
  k_cast_pad<<<(nXP + 255) / 256, 256, 0, stream>>>(x1, x1bf);
  k_cast_pad<<<(nXP + 255) / 256, 256, 0, stream>>>(x2, x2bf);
  k_mean_v<<<(kB * kF + 255) / 256, 256, 0, stream>>>(x1, x1m);
  // projections
  dim3 pg(kI / 64, kVp / 16, kB);
  k_proj_gemm<<<pg, 128, 0, stream>>>(wg, bg, x2bf, gb);
  k_proj_gemm<<<pg, 128, 0, stream>>>(wth, bth, x1bf, thb);
  k_proj_gemm<<<pg, 128, 0, stream>>>(wph, bph, x2bf, phb);
  // attention maps (fused GEMM + softmax)
  k_energy_time<<<dim3(kI / 16, kB), 256, 0, stream>>>(thb, phb, at);
  k_energy_space<<<kB, 256, 0, stream>>>(thb, phb, as);
  // z2 = attn_time @ g @ attn_space
  k_gemm_at_g<<<pg, 128, 0, stream>>>(at, gb, ybf);
  k_gemm_z2<<<pg, 128, 0, stream>>>(ybf, as, z2bf);
  // final W conv + BN + residual epilogue
  k_final<<<dim3(kF / 64, kVp / 16, kB), 128, 0, stream>>>(wW, z2bf, W_b, W_bn,
                                                           x2, x1m, out2);
  // output 1 = x1
  size_t n1 = (size_t)kB * kF * kV;
  k_copy<<<(unsigned)((n1 + 255) / 256), 256, 0, stream>>>(x1, out1, n1);
}
